// ProteinMPNN_19997367730448
// MI455X (gfx1250) — matrix operand order repaired and checked by reading, not compiled, not run on hardware
//
#include <hip/hip_runtime.h>
#include <hip/hip_bf16.h>
#include <math.h>

// ---------------- problem dims ----------------
#define BB   4
#define LL   2048
#define KK   48
#define HH   128
#define FFD  512
#define SCALE_INV (1.0f / 30.0f)
#define EPSV 1e-5f

// LDS strides (bf16 elements), padded, multiples of 8 (16B alignment of chunks)
#define ST3H 392   // 384 + 8
#define STH  136   // 128 + 8
#define STFF 520   // 512 + 8

typedef __attribute__((ext_vector_type(16))) __bf16 bf16x16;
typedef __attribute__((ext_vector_type(8)))  __bf16 bf16x8;
typedef __attribute__((ext_vector_type(8)))  float  f32x8;

// ---------------- WMMA helpers ----------------
__device__ __forceinline__ f32x8 wmma_bf16(bf16x16 a, bf16x16 b, f32x8 c) {
  // D = A(16x32 bf16) * B(32x16 bf16) + C(16x16 f32)
  return __builtin_amdgcn_wmma_f32_16x16x32_bf16(false, a, false, b, (short)0, c,
                                                 false, false);
}

// A fragment (16x32 bf16) from LDS, ISA layout:
// lanes 0-15: M=lane, elems 0-7 -> K=k0..k0+7, elems 8-15 -> K=k0+16..k0+23
// lanes 16-31: M=lane-16, elems 0-7 -> K=k0+8.., elems 8-15 -> K=k0+24..
__device__ __forceinline__ bf16x16 load_a_frag(const __bf16* lds, int row0, int k0,
                                               int stride, int lane) {
  int row = row0 + (lane & 15);
  int kb  = k0 + ((lane >> 4) << 3);
  const __bf16* p = lds + row * stride + kb;
  bf16x8 lo = *(const bf16x8*)p;
  bf16x8 hi = *(const bf16x8*)(p + 16);
  bf16x16 a;
#pragma unroll
  for (int i = 0; i < 8; ++i) { a[i] = lo[i]; a[i + 8] = hi[i]; }
  return a;
}

// B fragment: weights pre-packed in fragment order -> one contiguous 32B load
__device__ __forceinline__ bf16x16 load_b_frag(const __bf16* pk, int kt, int nt,
                                               int NT, int lane) {
  return *(const bf16x16*)(pk + (((size_t)(kt * NT + nt)) * 32 + lane) * 16);
}

__device__ __forceinline__ float gelu_f(float x) {
  return 0.5f * x * (1.0f + erff(x * 0.7071067811865476f));
}

// C tile (VGPR r: lanes0-15 M=r N=lane; lanes16-31 M=r+8 N=lane-16) -> LDS bf16
__device__ __forceinline__ void store_tile_act(__bf16* lds, f32x8 acc, int mrow0,
                                               int ncol0, int stride, float bias,
                                               int lane, bool do_gelu) {
  int col   = ncol0 + (lane & 15);
  int rbase = mrow0 + ((lane >> 4) << 3);
#pragma unroll
  for (int r = 0; r < 8; ++r) {
    float v = acc[r] + bias;
    if (do_gelu) v = gelu_f(v);
    lds[(rbase + r) * stride + col] = (__bf16)v;
  }
}

// ---------------- kernel 0: weight packing fp32 [Kd x N] -> bf16 B-fragments ----
// B layout (ISA 16-bit B, K=32): lane p<16: col=p, elems 0-7 -> K=kt*32+e, 8-15 -> K=kt*32+16+(e-8)
//                                lane p>=16: col=p-16, elems 0-7 -> K=kt*32+8+e, 8-15 -> K=kt*32+24+(e-8)
__global__ void pack_weights_kernel(const float* __restrict__ src,
                                    __bf16* __restrict__ dst, int Kd, int N) {
  int lane = threadIdx.x;            // 32 threads
  int NT   = N >> 4;
  int tile = blockIdx.x;             // kt*NT + nt
  int kt   = tile / NT;
  int nt   = tile - kt * NT;
  int n    = nt * 16 + (lane & 15);
  int kb   = kt * 32 + ((lane >> 4) << 3);
  __bf16* d = dst + ((size_t)tile * 32 + lane) * 16;
#pragma unroll
  for (int e = 0; e < 16; ++e)
    d[e] = (__bf16)src[(size_t)(kb + e + ((e >> 3) << 3)) * N + n];
}

// ---------------- kernel 1: node message MLP + K-reduce + LN1 -> hv1 ----------
__global__ __launch_bounds__(96) void node_message_kernel(
    const float* __restrict__ hV, const float* __restrict__ hE,
    const int* __restrict__ Eidx, const float* __restrict__ maskA,
    const __bf16* __restrict__ W1pk, const float* __restrict__ b1,
    const __bf16* __restrict__ W2pk, const float* __restrict__ b2,
    const __bf16* __restrict__ W3pk, const float* __restrict__ b3,
    const float* __restrict__ n1g, const float* __restrict__ n1b,
    float* __restrict__ hv1) {
  extern __shared__ char smem[];
  __bf16* A1   = (__bf16*)smem;                       // 48*392*2 = 37632
  __bf16* A2   = (__bf16*)(smem + 37632);             // 48*136*2 = 13056
  __bf16* A3   = (__bf16*)(smem + 37632 + 13056);     // 13056
  float*  outS = (float*)(smem + 63744);              // 128 f
  float*  mkS  = (float*)(smem + 64256);              // 48 f
  float*  red  = (float*)(smem + 64448);              // 32 f
  float*  stat = (float*)(smem + 64576);              // 2 f

  int tid = threadIdx.x, bid = blockIdx.x;
  int b = bid >> 11, l = bid & (LL - 1);
  const float* hVc = hV + ((size_t)(b * LL + l)) * HH;
  const float* hEc = hE + ((size_t)(b * LL + l)) * KK * HH;
  const int*   ec  = Eidx + (size_t)(b * LL + l) * KK;

  for (int i = tid; i < KK; i += 96) mkS[i] = maskA[(size_t)(b * LL + l) * KK + i];
  for (int i = tid; i < HH; i += 96) outS[i] = 0.f;

  // build concatenated [hV | hE_k | hV_nbr] rows as bf16
  for (int idx = tid; idx < KK * 3 * HH; idx += 96) {
    int row = idx / (3 * HH);
    int c   = idx - row * 3 * HH;
    float v;
    if (c < HH)           v = hVc[c];
    else if (c < 2 * HH)  v = hEc[row * HH + (c - HH)];
    else { int j = ec[row]; v = hV[((size_t)(b * LL + j)) * HH + (c - 2 * HH)]; }
    A1[row * ST3H + c] = (__bf16)v;
  }
  __syncthreads();

  int wave = tid >> 5, lane = tid & 31;
  int mrow0 = wave * 16;
  const f32x8 zf = {0.f, 0.f, 0.f, 0.f, 0.f, 0.f, 0.f, 0.f};

  { // GEMM1: 48x384 @ 384x128, gelu -> A2
    f32x8 acc[8];
#pragma unroll
    for (int nt = 0; nt < 8; ++nt) acc[nt] = zf;
    for (int kt = 0; kt < 12; ++kt) {
      bf16x16 a = load_a_frag(A1, mrow0, kt * 32, ST3H, lane);
#pragma unroll
      for (int nt = 0; nt < 8; ++nt)
        acc[nt] = wmma_bf16(a, load_b_frag(W1pk, kt, nt, 8, lane), acc[nt]);
    }
#pragma unroll
    for (int nt = 0; nt < 8; ++nt) {
      float bc = b1[nt * 16 + (lane & 15)];
      store_tile_act(A2, acc[nt], mrow0, nt * 16, STH, bc, lane, true);
    }
  }
  __syncthreads();
  { // GEMM2: 48x128 @ 128x128, gelu -> A3
    f32x8 acc[8];
#pragma unroll
    for (int nt = 0; nt < 8; ++nt) acc[nt] = zf;
    for (int kt = 0; kt < 4; ++kt) {
      bf16x16 a = load_a_frag(A2, mrow0, kt * 32, STH, lane);
#pragma unroll
      for (int nt = 0; nt < 8; ++nt)
        acc[nt] = wmma_bf16(a, load_b_frag(W2pk, kt, nt, 8, lane), acc[nt]);
    }
#pragma unroll
    for (int nt = 0; nt < 8; ++nt) {
      float bc = b2[nt * 16 + (lane & 15)];
      store_tile_act(A3, acc[nt], mrow0, nt * 16, STH, bc, lane, true);
    }
  }
  __syncthreads();
  { // GEMM3 + masked reduce over K rows
    f32x8 acc[8];
#pragma unroll
    for (int nt = 0; nt < 8; ++nt) acc[nt] = zf;
    for (int kt = 0; kt < 4; ++kt) {
      bf16x16 a = load_a_frag(A3, mrow0, kt * 32, STH, lane);
#pragma unroll
      for (int nt = 0; nt < 8; ++nt)
        acc[nt] = wmma_bf16(a, load_b_frag(W3pk, kt, nt, 8, lane), acc[nt]);
    }
    int rbase = mrow0 + ((lane >> 4) << 3);
#pragma unroll
    for (int nt = 0; nt < 8; ++nt) {
      int c = nt * 16 + (lane & 15);
      float bc = b3[c];
      float s = 0.f;
#pragma unroll
      for (int r = 0; r < 8; ++r) s += (acc[nt][r] + bc) * mkS[rbase + r];
      atomicAdd(&outS[c], s);
    }
  }
  __syncthreads();
  // residual + LN1
  for (int c = tid; c < HH; c += 96) outS[c] = hVc[c] + outS[c] * SCALE_INV;
  __syncthreads();
  if (tid < 32) {
    float s = 0.f;
#pragma unroll
    for (int i = 0; i < 4; ++i) s += outS[tid * 4 + i];
    red[tid] = s;
  }
  __syncthreads();
  if (tid == 0) {
    float s = 0.f;
    for (int i = 0; i < 32; ++i) s += red[i];
    stat[0] = s * (1.f / 128.f);
  }
  __syncthreads();
  float mean = stat[0];
  if (tid < 32) {
    float s = 0.f;
#pragma unroll
    for (int i = 0; i < 4; ++i) { float d = outS[tid * 4 + i] - mean; s += d * d; }
    red[tid] = s;
  }
  __syncthreads();
  if (tid == 0) {
    float s = 0.f;
    for (int i = 0; i < 32; ++i) s += red[i];
    stat[1] = s * (1.f / 128.f);
  }
  __syncthreads();
  float rs = rsqrtf(stat[1] + EPSV);
  float* dst = hv1 + ((size_t)(b * LL + l)) * HH;
  for (int c = tid; c < HH; c += 96)
    dst[c] = (outS[c] - mean) * rs * n1g[c] + n1b[c];
}

// ---------------- kernel 2: FFN + LN2 + mask -> out_hV, hv2 -------------------
__global__ __launch_bounds__(128) void ffn_kernel(
    const float* __restrict__ hv1, const float* __restrict__ maskV,
    const __bf16* __restrict__ Wginpk, const float* __restrict__ bgin,
    const __bf16* __restrict__ Wgoutpk, const float* __restrict__ bgout,
    const float* __restrict__ n2g, const float* __restrict__ n2b,
    float* __restrict__ out_hV, float* __restrict__ hv2) {
  extern __shared__ char smem[];
  __bf16* As  = (__bf16*)smem;                 // 16*136*2 = 4352
  __bf16* Mid = (__bf16*)(smem + 4352);        // 16*520*2 = 16640
  float*  Cs  = (float*)(smem + 20992);        // 16*136*4 = 8704

  int tid = threadIdx.x, bid = blockIdx.x;
  int b = bid >> 7, lt = bid & 127;
  int l0 = lt * 16;
  const float* src = hv1 + ((size_t)(b * LL + l0)) * HH;

  for (int idx = tid; idx < 16 * HH; idx += 128) {
    int r = idx >> 7, c = idx & 127;
    As[r * STH + c] = (__bf16)src[r * HH + c];
  }
  __syncthreads();

  int wave = tid >> 5, lane = tid & 31;
  const f32x8 zf = {0.f, 0.f, 0.f, 0.f, 0.f, 0.f, 0.f, 0.f};

  { // GEMM: 16x128 @ 128x512, gelu -> Mid ; wave owns 8 of 32 N-tiles
    f32x8 acc[8];
#pragma unroll
    for (int j = 0; j < 8; ++j) acc[j] = zf;
    for (int kt = 0; kt < 4; ++kt) {
      bf16x16 a = load_a_frag(As, 0, kt * 32, STH, lane);
#pragma unroll
      for (int j = 0; j < 8; ++j)
        acc[j] = wmma_bf16(a, load_b_frag(Wginpk, kt, wave * 8 + j, 32, lane), acc[j]);
    }
#pragma unroll
    for (int j = 0; j < 8; ++j) {
      int c = (wave * 8 + j) * 16 + (lane & 15);
      store_tile_act(Mid, acc[j], 0, (wave * 8 + j) * 16, STFF, bgin[c], lane, true);
    }
  }
  __syncthreads();
  { // GEMM: 16x512 @ 512x128 ; wave owns 2 of 8 N-tiles; + residual -> Cs
    f32x8 acc[2];
    acc[0] = zf; acc[1] = zf;
    for (int kt = 0; kt < 16; ++kt) {
      bf16x16 a = load_a_frag(Mid, 0, kt * 32, STFF, lane);
#pragma unroll
      for (int j = 0; j < 2; ++j)
        acc[j] = wmma_bf16(a, load_b_frag(Wgoutpk, kt, wave * 2 + j, 8, lane), acc[j]);
    }
    int rbase = (lane >> 4) << 3;
#pragma unroll
    for (int j = 0; j < 2; ++j) {
      int c = (wave * 2 + j) * 16 + (lane & 15);
      float bc = bgout[c];
#pragma unroll
      for (int r = 0; r < 8; ++r) {
        int row = rbase + r;
        Cs[row * STH + c] = acc[j][r] + bc + src[row * HH + c];
      }
    }
  }
  __syncthreads();
  // per-row LN2 + mask
  if (tid < 16) {
    int row = tid, l = l0 + row;
    float m = 0.f;
    for (int c = 0; c < HH; ++c) m += Cs[row * STH + c];
    m *= (1.f / 128.f);
    float v = 0.f;
    for (int c = 0; c < HH; ++c) { float d = Cs[row * STH + c] - m; v += d * d; }
    v *= (1.f / 128.f);
    float rsv = rsqrtf(v + EPSV);
    float mk = maskV[(size_t)b * LL + l];
    float* o1 = out_hV + ((size_t)(b * LL + l)) * HH;
    float* o2 = hv2 + ((size_t)(b * LL + l)) * HH;
    for (int c = 0; c < HH; ++c) {
      float y = mk * ((Cs[row * STH + c] - m) * rsv * n2g[c] + n2b[c]);
      o1[c] = y; o2[c] = y;
    }
  }
}

// ---------------- kernel 3: edge message MLP + LN3 -> out_hE ------------------
__global__ __launch_bounds__(96) void edge_message_kernel(
    const float* __restrict__ hv2, const float* __restrict__ hE,
    const int* __restrict__ Eidx,
    const __bf16* __restrict__ W11pk, const float* __restrict__ b11,
    const __bf16* __restrict__ W12pk, const float* __restrict__ b12,
    const __bf16* __restrict__ W13pk, const float* __restrict__ b13,
    const float* __restrict__ n3g, const float* __restrict__ n3b,
    float* __restrict__ out_hE) {
  extern __shared__ char smem[];
  __bf16* A1 = (__bf16*)smem;                       // 37632
  __bf16* A2 = (__bf16*)(smem + 37632);             // 13056
  __bf16* A3 = (__bf16*)(smem + 50688);             // 13056
  float*  Cs = (float*)(smem + 63744);              // 48*136*4 = 26112

  int tid = threadIdx.x, bid = blockIdx.x;
  int b = bid >> 11, l = bid & (LL - 1);
  const float* hVc = hv2 + ((size_t)(b * LL + l)) * HH;
  const float* hEc = hE + ((size_t)(b * LL + l)) * KK * HH;
  const int*   ec  = Eidx + (size_t)(b * LL + l) * KK;

  for (int idx = tid; idx < KK * 3 * HH; idx += 96) {
    int row = idx / (3 * HH);
    int c   = idx - row * 3 * HH;
    float v;
    if (c < HH)           v = hVc[c];
    else if (c < 2 * HH)  v = hEc[row * HH + (c - HH)];
    else { int j = ec[row]; v = hv2[((size_t)(b * LL + j)) * HH + (c - 2 * HH)]; }
    A1[row * ST3H + c] = (__bf16)v;
  }
  __syncthreads();

  int wave = tid >> 5, lane = tid & 31;
  int mrow0 = wave * 16;
  const f32x8 zf = {0.f, 0.f, 0.f, 0.f, 0.f, 0.f, 0.f, 0.f};

  { // GEMM1 -> A2
    f32x8 acc[8];
#pragma unroll
    for (int nt = 0; nt < 8; ++nt) acc[nt] = zf;
    for (int kt = 0; kt < 12; ++kt) {
      bf16x16 a = load_a_frag(A1, mrow0, kt * 32, ST3H, lane);
#pragma unroll
      for (int nt = 0; nt < 8; ++nt)
        acc[nt] = wmma_bf16(a, load_b_frag(W11pk, kt, nt, 8, lane), acc[nt]);
    }
#pragma unroll
    for (int nt = 0; nt < 8; ++nt)
      store_tile_act(A2, acc[nt], mrow0, nt * 16, STH, b11[nt * 16 + (lane & 15)],
                     lane, true);
  }
  __syncthreads();
  { // GEMM2 -> A3
    f32x8 acc[8];
#pragma unroll
    for (int nt = 0; nt < 8; ++nt) acc[nt] = zf;
    for (int kt = 0; kt < 4; ++kt) {
      bf16x16 a = load_a_frag(A2, mrow0, kt * 32, STH, lane);
#pragma unroll
      for (int nt = 0; nt < 8; ++nt)
        acc[nt] = wmma_bf16(a, load_b_frag(W12pk, kt, nt, 8, lane), acc[nt]);
    }
#pragma unroll
    for (int nt = 0; nt < 8; ++nt)
      store_tile_act(A3, acc[nt], mrow0, nt * 16, STH, b12[nt * 16 + (lane & 15)],
                     lane, true);
  }
  __syncthreads();
  { // GEMM3 -> Cs (fp32)
    f32x8 acc[8];
#pragma unroll
    for (int nt = 0; nt < 8; ++nt) acc[nt] = zf;
    for (int kt = 0; kt < 4; ++kt) {
      bf16x16 a = load_a_frag(A3, mrow0, kt * 32, STH, lane);
#pragma unroll
      for (int nt = 0; nt < 8; ++nt)
        acc[nt] = wmma_bf16(a, load_b_frag(W13pk, kt, nt, 8, lane), acc[nt]);
    }
    int rbase = mrow0 + ((lane >> 4) << 3);
#pragma unroll
    for (int nt = 0; nt < 8; ++nt) {
      int c = nt * 16 + (lane & 15);
      float bc = b13[c];
#pragma unroll
      for (int r = 0; r < 8; ++r) Cs[(rbase + r) * STH + c] = acc[nt][r] + bc;
    }
  }
  __syncthreads();
  // per-edge-row residual + LN3
  if (tid < KK) {
    int k = tid;
    const float* er = hEc + (size_t)k * HH;
    float* orow = out_hE + (((size_t)(b * LL + l)) * KK + k) * HH;
    float m = 0.f;
    for (int c = 0; c < HH; ++c) {
      float x = er[c] + Cs[k * STH + c];
      Cs[k * STH + c] = x;
      m += x;
    }
    m *= (1.f / 128.f);
    float v = 0.f;
    for (int c = 0; c < HH; ++c) { float d = Cs[k * STH + c] - m; v += d * d; }
    v *= (1.f / 128.f);
    float rsv = rsqrtf(v + EPSV);
    for (int c = 0; c < HH; ++c)
      orow[c] = (Cs[k * STH + c] - m) * rsv * n3g[c] + n3b[c];
  }
}

// ---------------- workspace layout (bytes) ----------------
#define OFF_W1    0
#define OFF_W2    98304
#define OFF_W3    131072
#define OFF_W11   163840
#define OFF_W12   262144
#define OFF_W13   294912
#define OFF_WGIN  327680
#define OFF_WGOUT 458752
#define OFF_HV1   589824
#define OFF_HV2   4784128
// total 8,978,432 bytes

#define SMEM1 64584
#define SMEM2 29696
#define SMEM3 89856

extern "C" void kernel_launch(void* const* d_in, const int* in_sizes, int n_in,
                              void* d_out, int out_size, void* d_ws, size_t ws_size,
                              hipStream_t stream) {
  const float* hV    = (const float*)d_in[0];
  const float* hE    = (const float*)d_in[1];
  const int*   Eidx  = (const int*)d_in[2];
  const float* maskV = (const float*)d_in[3];
  const float* maskA = (const float*)d_in[4];
  const float* W1  = (const float*)d_in[5];  const float* b1  = (const float*)d_in[6];
  const float* W2  = (const float*)d_in[7];  const float* b2  = (const float*)d_in[8];
  const float* W3  = (const float*)d_in[9];  const float* b3  = (const float*)d_in[10];
  const float* W11 = (const float*)d_in[11]; const float* b11 = (const float*)d_in[12];
  const float* W12 = (const float*)d_in[13]; const float* b12 = (const float*)d_in[14];
  const float* W13 = (const float*)d_in[15]; const float* b13 = (const float*)d_in[16];
  const float* Wgin  = (const float*)d_in[17]; const float* bgin  = (const float*)d_in[18];
  const float* Wgout = (const float*)d_in[19]; const float* bgout = (const float*)d_in[20];
  const float* n1g = (const float*)d_in[21]; const float* n1b = (const float*)d_in[22];
  const float* n2g = (const float*)d_in[23]; const float* n2b = (const float*)d_in[24];
  const float* n3g = (const float*)d_in[25]; const float* n3b = (const float*)d_in[26];

  char* ws = (char*)d_ws;
  __bf16* W1pk    = (__bf16*)(ws + OFF_W1);
  __bf16* W2pk    = (__bf16*)(ws + OFF_W2);
  __bf16* W3pk    = (__bf16*)(ws + OFF_W3);
  __bf16* W11pk   = (__bf16*)(ws + OFF_W11);
  __bf16* W12pk   = (__bf16*)(ws + OFF_W12);
  __bf16* W13pk   = (__bf16*)(ws + OFF_W13);
  __bf16* Wginpk  = (__bf16*)(ws + OFF_WGIN);
  __bf16* Wgoutpk = (__bf16*)(ws + OFF_WGOUT);
  float*  hv1     = (float*)(ws + OFF_HV1);
  float*  hv2     = (float*)(ws + OFF_HV2);

  float* out_hV = (float*)d_out;
  float* out_hE = out_hV + (size_t)BB * LL * HH;

  // pack all weight matrices to bf16 WMMA fragment order
  pack_weights_kernel<<<(384/32)*(128/16), 32, 0, stream>>>(W1,  W1pk,  384, 128);
  pack_weights_kernel<<<(128/32)*(128/16), 32, 0, stream>>>(W2,  W2pk,  128, 128);
  pack_weights_kernel<<<(128/32)*(128/16), 32, 0, stream>>>(W3,  W3pk,  128, 128);
  pack_weights_kernel<<<(384/32)*(128/16), 32, 0, stream>>>(W11, W11pk, 384, 128);
  pack_weights_kernel<<<(128/32)*(128/16), 32, 0, stream>>>(W12, W12pk, 128, 128);
  pack_weights_kernel<<<(128/32)*(128/16), 32, 0, stream>>>(W13, W13pk, 128, 128);
  pack_weights_kernel<<<(128/32)*(512/16), 32, 0, stream>>>(Wgin,  Wginpk,  128, 512);
  pack_weights_kernel<<<(512/32)*(128/16), 32, 0, stream>>>(Wgout, Wgoutpk, 512, 128);

  node_message_kernel<<<BB * LL, 96, SMEM1, stream>>>(
      hV, hE, Eidx, maskA, W1pk, b1, W2pk, b2, W3pk, b3, n1g, n1b, hv1);

  ffn_kernel<<<BB * (LL / 16), 128, SMEM2, stream>>>(
      hv1, maskV, Wginpk, bgin, Wgoutpk, bgout, n2g, n2b, out_hV, hv2);

  edge_message_kernel<<<BB * LL, 96, SMEM3, stream>>>(
      hv2, hE, Eidx, W11pk, b11, W12pk, b12, W13pk, b13, n3g, n3b, out_hE);
}
